// RNN_10084583211558
// MI455X (gfx1250) — matrix operand order repaired and checked by reading, not compiled, not run on hardware
//
#include <hip/hip_runtime.h>
#include <math.h>

// ---------------- problem constants ----------------
#define BB    4
#define SS    8192
#define DD    512          // DIN == DH == DOUT == 512
#define MTOT  (BB * SS)    // 32768 rows in the flattened GEMMs
#define CHUNK 128
#define NCH   (SS / CHUNK) // 64 chunks per sequence
#define GATE_MIN 6.25e-06f // 1/(4*200^2)
#define KSTEP 32
#define NKS   (DD / KSTEP) // 16 K-steps per 512-K GEMM

// ---------------- WMMA vector types ----------------
typedef __attribute__((ext_vector_type(16))) __bf16       v16bf;
typedef __attribute__((ext_vector_type(8)))  float        v8f;
typedef __attribute__((ext_vector_type(4)))  unsigned int uint4v;
typedef __attribute__((ext_vector_type(4)))  int          int4v;

union FragBF {
  v16bf v;
  uint4v u[2];
};

// float -> bf16 round-to-nearest-even (bit manipulation)
__device__ __forceinline__ __bf16 f2bf(float f) {
  union { float f; unsigned u; } x; x.f = f;
  unsigned r = (x.u + 0x7FFFu + ((x.u >> 16) & 1u)) >> 16;
  union { unsigned short s; __bf16 b; } y; y.s = (unsigned short)r;
  return y.b;
}

__device__ __forceinline__ float sigmoidf_dev(float z) {
  return 1.0f / (1.0f + __expf(-z));
}

// ---------------- async global -> LDS copy (16B), with fallback ----------------
__device__ __forceinline__ void async_copy16(const __bf16* g, __bf16* l) {
#if defined(__has_builtin) && __has_builtin(__builtin_amdgcn_global_load_async_to_lds_b128)
  __builtin_amdgcn_global_load_async_to_lds_b128(
      (__attribute__((address_space(1))) int4v*)g,
      (__attribute__((address_space(3))) int4v*)l, 0, 0);
#else
  *(uint4v*)l = *(const uint4v*)g;   // vmem load + ds_store fallback
#endif
}

template <int N>
__device__ __forceinline__ void async_wait_le() {
#if defined(__has_builtin) && __has_builtin(__builtin_amdgcn_s_wait_asynccnt)
  __builtin_amdgcn_s_wait_asynccnt((short)N);
#elif defined(__HIP_DEVICE_COMPILE__)
  asm volatile("s_wait_asynccnt %0" ::"i"(N));
#endif
}

// ---- A fragment: 16x32 bf16, row-major source [M][512] ----
// lane 0-15 : row m0+lane,    K = kb+{0..7} (elems 0..7), kb+{16..23} (elems 8..15)
// lane 16-31: row m0+lane-16, K = kb+{8..15},             kb+{24..31}
__device__ __forceinline__ v16bf load_frag_A(const __bf16* A, int m0, int kb, int lane) {
  int m    = m0 + (lane & 15);
  int koff = kb + ((lane >> 4) << 3);            // +0 or +8
  const __bf16* p = A + (size_t)m * DD + koff;
  FragBF f;
  f.u[0] = *(const uint4v*)(p);
  f.u[1] = *(const uint4v*)(p + 16);
  return f.v;
}

// ---- B fragment from LDS-staged 512x32 panel, chunk-swizzled ----
// panel row n: 32 bf16 (64B) = 4 chunks of 16B; swizzle c' = c ^ ((n>>2)&3)
// lane = column n0+(lane&15); lanes 0-15 need chunks {0,2}, lanes 16-31 chunks {1,3}
__device__ __forceinline__ v16bf load_frag_B_lds(const __bf16* buf, int n0, int lane) {
  int nn   = n0 + (lane & 15);
  int half = lane >> 4;
  int sw   = (nn >> 2) & 3;
  const __bf16* row = buf + nn * KSTEP;
  FragBF f;
  f.u[0] = *(const uint4v*)(row + (((half)     ^ sw) << 3));
  f.u[1] = *(const uint4v*)(row + (((half + 2) ^ sw) << 3));
  return f.v;
}

// ---- stage one 512x32 weight slice (32KB) into an LDS buffer ----
// 2048 chunks of 16B, 8 per thread (256 threads)
__device__ __forceinline__ void stage_panel(__bf16* buf, const __bf16* WT, int kb, int tid) {
#pragma unroll
  for (int i = 0; i < 8; ++i) {
    int g = i * 256 + tid;             // chunk id
    int n = g >> 2;
    int c = g & 3;
    int sw = (n >> 2) & 3;
    const __bf16* src = WT + (size_t)n * DD + kb + (c << 3);
    __bf16* dst = buf + n * KSTEP + ((c ^ sw) << 3);
    async_copy16(src, dst);
  }
}

// =====================================================================
// Kernel 0: repack W (f32, [k][n]) -> WT (bf16, [n][k]) for all 3 weights
// =====================================================================
__global__ void repack_weights(const float* __restrict__ Wdx,
                               const float* __restrict__ Win,
                               const float* __restrict__ Wout,
                               __bf16* __restrict__ WTdx,
                               __bf16* __restrict__ WTin,
                               __bf16* __restrict__ WTout) {
  int id = blockIdx.x * blockDim.x + threadIdx.x;   // 3*512*512 threads
  int which = id >> 18;
  int rem   = id & ((1 << 18) - 1);
  int k = rem >> 9;
  int n = rem & 511;
  const float* src = (which == 0) ? Wdx : (which == 1) ? Win : Wout;
  __bf16* dst      = (which == 0) ? WTdx : (which == 1) ? WTin : WTout;
  dst[(size_t)n * DD + k] = f2bf(src[(size_t)k * DD + n]);
}

// =====================================================================
// Kernel 1: x f32 -> bf16 (same layout)
// =====================================================================
__global__ void convert_x(const float* __restrict__ x, __bf16* __restrict__ xbf) {
  size_t id = (size_t)blockIdx.x * blockDim.x + threadIdx.x;
  xbf[id] = f2bf(x[id]);
}

// =====================================================================
// Kernel 2: xh = gate * (x @ W_in + b_in)
// block = 256 thr (8 waves); block tile 32M x 512N, wave tile 16M x 128N
// B panel double-buffered in LDS via async global->LDS copies
// =====================================================================
__global__ void gemm_xh(const __bf16* __restrict__ xbf,
                        const __bf16* __restrict__ WTin,
                        const float* __restrict__ a_logit,
                        const float* __restrict__ b_in,
                        float* __restrict__ xh) {
  __shared__ __bf16 Bs[2][DD * KSTEP];   // 2 x 32KB
  int tid  = threadIdx.x;
  int wave = tid >> 5;
  int lane = tid & 31;
  int mi = wave & 1;          // 2 M-tiles per block
  int ni = wave >> 1;         // 4 N-groups of 128
  int m0 = blockIdx.x * 32 + mi * 16;

  v8f acc[8];
#pragma unroll
  for (int j = 0; j < 8; ++j) acc[j] = (v8f){0.f,0.f,0.f,0.f,0.f,0.f,0.f,0.f};

  stage_panel(Bs[0], WTin, 0, tid);
  for (int s = 0; s < NKS; ++s) {
    int kb = s << 5;
    v16bf a = load_frag_A(xbf, m0, kb, lane);
    if (s + 1 < NKS) {
      stage_panel(Bs[(s + 1) & 1], WTin, kb + KSTEP, tid);
      async_wait_le<8>();                // previous batch (8/thread) complete
    } else {
      async_wait_le<0>();
    }
    __syncthreads();
    const __bf16* buf = Bs[s & 1];
#pragma unroll
    for (int j = 0; j < 8; ++j) {
      v16bf b = load_frag_B_lds(buf, ni * 128 + j * 16, lane);
      acc[j] = __builtin_amdgcn_wmma_f32_16x16x32_bf16(
          false, a, false, b, (short)0, acc[j], false, false);
    }
    __syncthreads();                     // buffer reuse protection
  }

  // epilogue: apply bias then gate, store f32
  int lane_n = lane & 15;
  int mbase  = m0 + ((lane >> 4) << 3);  // +0 or +8
#pragma unroll
  for (int j = 0; j < 8; ++j) {
    int n = ni * 128 + j * 16 + lane_n;
    float av   = sigmoidf_dev(a_logit[n]);
    float gate = __fsqrt_rn(fmaxf(1.0f - av * av, GATE_MIN));
    float bias = b_in[n];
#pragma unroll
    for (int r = 0; r < 8; ++r) {
      int m = mbase + r;
      xh[(size_t)m * DD + n] = gate * (acc[j][r] + bias);
    }
  }
}

// =====================================================================
// Kernel 3: local inclusive scan (in place) over chunks of 128 steps
// =====================================================================
__global__ void scan_local(float* __restrict__ xh,
                           float* __restrict__ chunk_last,
                           const float* __restrict__ a_logit) {
  int bx = blockIdx.x;
  int c     = (bx & 1) * 256 + threadIdx.x;
  int chunk = (bx >> 1) & (NCH - 1);
  int b     = bx >> 7;
  float a = sigmoidf_dev(a_logit[c]);
  size_t base = ((size_t)b * SS + (size_t)chunk * CHUNK) * DD + c;
  float h = 0.0f;
#pragma unroll 4
  for (int t = 0; t < CHUNK; ++t) {
    h = __fmaf_rn(a, h, xh[base + (size_t)t * DD]);
    xh[base + (size_t)t * DD] = h;
  }
  chunk_last[((size_t)b * NCH + chunk) * DD + c] = h;
}

// =====================================================================
// Kernel 4: carry scan across the 64 chunks (2048 threads total)
// =====================================================================
__global__ void scan_carry(const float* __restrict__ chunk_last,
                           float* __restrict__ carry_buf,
                           const float* __restrict__ a_logit) {
  int idx = blockIdx.x * blockDim.x + threadIdx.x;  // 0 .. B*DD-1
  int b = idx >> 9;
  int c = idx & 511;
  float a = sigmoidf_dev(a_logit[c]);
  float aL = a;
#pragma unroll
  for (int i = 0; i < 7; ++i) aL = aL * aL;         // a^128
  float carry = 0.0f;
  for (int j = 0; j < NCH; ++j) {
    size_t off = ((size_t)b * NCH + j) * DD + c;
    carry_buf[off] = carry;
    carry = __fmaf_rn(aL, carry, chunk_last[off]);
  }
}

// =====================================================================
// Kernel 5: apply carries, emit h as bf16 (+ h_last f32 to output tail)
// =====================================================================
__global__ void scan_apply(const float* __restrict__ xh,
                           const float* __restrict__ carry_buf,
                           const float* __restrict__ a_logit,
                           __bf16* __restrict__ hbf,
                           float* __restrict__ h_last) {
  int bx = blockIdx.x;
  int c     = (bx & 1) * 256 + threadIdx.x;
  int chunk = (bx >> 1) & (NCH - 1);
  int b     = bx >> 7;
  float a = sigmoidf_dev(a_logit[c]);
  float carry = carry_buf[((size_t)b * NCH + chunk) * DD + c];
  size_t base = ((size_t)b * SS + (size_t)chunk * CHUNK) * DD + c;
  float p = a;
  float hv = 0.0f;
#pragma unroll 4
  for (int t = 0; t < CHUNK; ++t) {
    hv = __fmaf_rn(p, carry, xh[base + (size_t)t * DD]);
    p *= a;
    hbf[base + (size_t)t * DD] = f2bf(hv);
  }
  if (chunk == NCH - 1) h_last[b * DD + c] = hv;
}

// =====================================================================
// Kernel 6: y = 0.5 * (x@W_dx + h@W_out + b_dx + b_out)  (two fused GEMMs)
// 32 K-steps: 0-15 stage WTdx / A=x, 16-31 stage WTout / A=h
// =====================================================================
__global__ void gemm_y(const __bf16* __restrict__ xbf,
                       const __bf16* __restrict__ hbf,
                       const __bf16* __restrict__ WTdx,
                       const __bf16* __restrict__ WTout,
                       const float* __restrict__ b_dx,
                       const float* __restrict__ b_out,
                       float* __restrict__ y) {
  __shared__ __bf16 Bs[2][DD * KSTEP];   // 2 x 32KB
  int tid  = threadIdx.x;
  int wave = tid >> 5;
  int lane = tid & 31;
  int mi = wave & 1;
  int ni = wave >> 1;
  int m0 = blockIdx.x * 32 + mi * 16;

  v8f acc[8];
#pragma unroll
  for (int j = 0; j < 8; ++j) acc[j] = (v8f){0.f,0.f,0.f,0.f,0.f,0.f,0.f,0.f};

  stage_panel(Bs[0], WTdx, 0, tid);
  for (int s = 0; s < 2 * NKS; ++s) {
    int kb = (s & (NKS - 1)) << 5;
    const __bf16* Asrc = (s < NKS) ? xbf : hbf;
    v16bf a = load_frag_A(Asrc, m0, kb, lane);
    if (s + 1 < 2 * NKS) {
      int s1 = s + 1;
      const __bf16* Wn = (s1 < NKS) ? WTdx : WTout;
      stage_panel(Bs[s1 & 1], Wn, (s1 & (NKS - 1)) << 5, tid);
      async_wait_le<8>();
    } else {
      async_wait_le<0>();
    }
    __syncthreads();
    const __bf16* buf = Bs[s & 1];
#pragma unroll
    for (int j = 0; j < 8; ++j) {
      v16bf b = load_frag_B_lds(buf, ni * 128 + j * 16, lane);
      acc[j] = __builtin_amdgcn_wmma_f32_16x16x32_bf16(
          false, a, false, b, (short)0, acc[j], false, false);
    }
    __syncthreads();
  }

  int lane_n = lane & 15;
  int mbase  = m0 + ((lane >> 4) << 3);
#pragma unroll
  for (int j = 0; j < 8; ++j) {
    int n = ni * 128 + j * 16 + lane_n;
    float bias = b_dx[n] + b_out[n];
#pragma unroll
    for (int r = 0; r < 8; ++r) {
      int m = mbase + r;
      y[(size_t)m * DD + n] = 0.5f * (acc[j][r] + bias);
    }
  }
}

// =====================================================================
// Host-side launcher
// =====================================================================
extern "C" void kernel_launch(void* const* d_in, const int* in_sizes, int n_in,
                              void* d_out, int out_size, void* d_ws, size_t ws_size,
                              hipStream_t stream) {
  const float* x       = (const float*)d_in[0];
  const float* a_logit = (const float*)d_in[1];
  const float* W_dx    = (const float*)d_in[2];
  const float* b_dx    = (const float*)d_in[3];
  const float* W_in    = (const float*)d_in[4];
  const float* b_in    = (const float*)d_in[5];
  const float* W_out   = (const float*)d_in[6];
  const float* b_out   = (const float*)d_in[7];

  float* out    = (float*)d_out;
  float* y_out  = out;                              // B*S*DD floats
  float* h_last = out + (size_t)MTOT * DD;          // B*DD floats

  char* w = (char*)d_ws;
  __bf16* WTdx  = (__bf16*)w;  w += (size_t)DD * DD * sizeof(__bf16);
  __bf16* WTin  = (__bf16*)w;  w += (size_t)DD * DD * sizeof(__bf16);
  __bf16* WTout = (__bf16*)w;  w += (size_t)DD * DD * sizeof(__bf16);
  __bf16* xbf   = (__bf16*)w;  w += (size_t)MTOT * DD * sizeof(__bf16);
  float*  xh    = (float*)w;   w += (size_t)MTOT * DD * sizeof(float);
  __bf16* hbf   = (__bf16*)w;  w += (size_t)MTOT * DD * sizeof(__bf16);
  float*  clast = (float*)w;   w += (size_t)BB * NCH * DD * sizeof(float);
  float*  carry = (float*)w;   w += (size_t)BB * NCH * DD * sizeof(float);

  // 0) repack weights: 3 * 512 * 512 elements
  repack_weights<<<(3 * DD * DD) / 256, 256, 0, stream>>>(
      W_dx, W_in, W_out, WTdx, WTin, WTout);

  // 1) x -> bf16
  convert_x<<<(MTOT * DD) / 256, 256, 0, stream>>>(x, xbf);

  // 2) xh = gate * (x @ W_in + b_in)
  gemm_xh<<<MTOT / 32, 256, 0, stream>>>(xbf, WTin, a_logit, b_in, xh);

  // 3) local scan per 128-chunk (in place)
  scan_local<<<BB * NCH * 2, 256, 0, stream>>>(xh, clast, a_logit);

  // 4) carry scan across chunks
  scan_carry<<<(BB * DD) / 256, 256, 0, stream>>>(clast, carry, a_logit);

  // 5) apply carries -> h (bf16) + h_last (f32)
  scan_apply<<<BB * NCH * 2, 256, 0, stream>>>(xh, carry, a_logit, hbf, h_last);

  // 6) y = 0.5 * (x @ W_dx + h @ W_out + biases)
  gemm_y<<<MTOT / 32, 256, 0, stream>>>(xbf, hbf, WTdx, WTout, b_dx, b_out, y_out);
}
